// SpatialGatingUnit_22187801051233
// MI455X (gfx1250) — matrix operand order repaired and checked by reading, not compiled, not run on hardware
//
#include <hip/hip_runtime.h>
#include <hip/hip_bf16.h>
#include <cstdint>

#define TLEN 243
#define DIMF 512
#define CB   256
#define FREQ 122

typedef __bf16  v16bf __attribute__((ext_vector_type(16)));
typedef float   v8f   __attribute__((ext_vector_type(8)));
typedef unsigned short v16us __attribute__((ext_vector_type(16)));

union Frag {
  unsigned d[8];
  v16us us;
  v16bf bf;
};

__device__ __forceinline__ unsigned short f2bf(float f) {
  unsigned u = __float_as_uint(f);
  u += 0x7FFFu + ((u >> 16) & 1u);   // round-to-nearest-even
  return (unsigned short)(u >> 16);
}

__device__ __forceinline__ float wave_sum(float v) {
#pragma unroll
  for (int m = 16; m >= 1; m >>= 1) v += __shfl_xor(v, m, 32);
  return v;
}

// ---------------------------------------------------------------------------
// P[f,s] = (1/sqrt(T)) * sum_k lf[f,k] * e^{-2*pi*i*k*s/T}
// ---------------------------------------------------------------------------
__global__ void build_P(const float* __restrict__ lf, float* __restrict__ P) {
  int idx = blockIdx.x * blockDim.x + threadIdx.x;
  if (idx >= FREQ * TLEN) return;
  int f = idx / TLEN, s = idx - f * TLEN;
  const float twoPiOverT = 6.28318530717958647692f / (float)TLEN;
  float re = 0.f, im = 0.f;
  for (int k = 0; k < FREQ; ++k) {
    float a = lf[(f * FREQ + k) * 2 + 0];
    float b = lf[(f * FREQ + k) * 2 + 1];
    int m = (k * s) % TLEN;            // exact integer range reduction
    float ang = -twoPiOverT * (float)m;
    float sn, cs; sincosf(ang, &sn, &cs);
    re += a * cs - b * sn;
    im += a * sn + b * cs;
  }
  const float invSqrtT = 0.0641500299099584f;
  P[idx * 2 + 0] = re * invSqrtT;
  P[idx * 2 + 1] = im * invSqrtT;
}

// ---------------------------------------------------------------------------
// M[t,s] = (1/sqrt(T)) * ( Re P[0,s] + 2*sum_{f>=1} Re(P[f,s]*e^{+2 pi i f t/T}) )
// stored bf16, zero-padded to 256x256
// ---------------------------------------------------------------------------
__global__ void build_M(const float* __restrict__ P, unsigned short* __restrict__ Mbf) {
  int idx = blockIdx.x * blockDim.x + threadIdx.x;   // 256*256
  int t = idx >> 8, s = idx & 255;
  float val = 0.f;
  if (t < TLEN && s < TLEN) {
    const float twoPiOverT = 6.28318530717958647692f / (float)TLEN;
    float acc = P[(0 * TLEN + s) * 2 + 0];           // Re P[0,s] (Im discarded by irfft)
    for (int f = 1; f < FREQ; ++f) {
      int m = (f * t) % TLEN;
      float ang = twoPiOverT * (float)m;
      float sn, cs; sincosf(ang, &sn, &cs);
      float pr = P[(f * TLEN + s) * 2 + 0];
      float pi = P[(f * TLEN + s) * 2 + 1];
      acc += 2.f * (pr * cs - pi * sn);
    }
    val = acc * 0.0641500299099584f;
  }
  Mbf[idx] = f2bf(val);
}

// ---------------------------------------------------------------------------
// Weights -> bf16, K-major per output channel.
// w1 flat index o*384 + (i*3+k) == im2col K index kappa -> identity copy.
// w2 flat index o*192 + (i*3+k) likewise.
// ---------------------------------------------------------------------------
__global__ void build_W(const float* __restrict__ w1, const float* __restrict__ w2,
                        unsigned short* __restrict__ B1t, unsigned short* __restrict__ B2t) {
  int idx = blockIdx.x * blockDim.x + threadIdx.x;
  if (idx < 128 * 384)                 B1t[idx] = f2bf(w1[idx]);
  else if (idx < 128 * 384 + 256 * 192) B2t[idx - 128 * 384] = f2bf(w2[idx - 128 * 384]);
}

// ---------------------------------------------------------------------------
// Spectral branch: per batch, out[t,c] (c<256) = sum_s M[t,s] * x[b,s,c]
// block = (stripe of 128 M-rows) x batch; 8 waves, each 8 Mtiles x 2 Ntiles.
// ---------------------------------------------------------------------------
__global__ __launch_bounds__(256) void fft_gemm(const float* __restrict__ x,
                                                const unsigned short* __restrict__ Mbf,
                                                float* __restrict__ out) {
  __shared__ unsigned short ldsA[128 * 32];   // [row][k] bf16
  __shared__ unsigned short ldsB[256 * 32];   // [c][k]  bf16 (transposed for frag loads)
  const int b = blockIdx.y;
  const int mbase = blockIdx.x * 128;
  const int tid = threadIdx.x;
  const int w = tid >> 5, lane = tid & 31;
  const int hl = lane >> 4, ln = lane & 15;
  const int n0 = w * 2;

  v8f zero = {};
  v8f acc[8][2];
#pragma unroll
  for (int i = 0; i < 8; ++i) { acc[i][0] = zero; acc[i][1] = zero; }

  for (int kc = 0; kc < 8; ++kc) {
    const int kb = kc * 32;
    __syncthreads();
    {   // stage A: Mbf[mbase+row][kb+k] -> ldsA
      const unsigned* src = (const unsigned*)Mbf;
#pragma unroll
      for (int it = 0; it < 8; ++it) {
        int dw = tid + it * 256;            // 2048 dwords
        int row = dw >> 4;
        int kp = dw & 15;
        ((unsigned*)ldsA)[dw] = src[(mbase + row) * 128 + (kb >> 1) + kp];
      }
    }
    {   // stage B: x[b, s=kb+k, c] -> bf16 transposed
      const int c = tid;
#pragma unroll 4
      for (int k = 0; k < 32; ++k) {
        int s = kb + k;
        float v = (s < TLEN) ? x[((size_t)b * TLEN + s) * DIMF + c] : 0.f;
        ldsB[c * 32 + k] = f2bf(v);
      }
    }
    __syncthreads();

    Frag bf0, bf1;
    {
      const unsigned* pb = (const unsigned*)ldsB;
      int c0 = n0 * 16 + ln;
      int c1 = c0 + 16;
      int off = hl * 8;
#pragma unroll
      for (int p = 0; p < 8; ++p) {
        bf0.d[p] = pb[c0 * 16 + off + p];
        bf1.d[p] = pb[c1 * 16 + off + p];
      }
    }
#pragma unroll
    for (int mt = 0; mt < 8; ++mt) {
      Frag af;
      const unsigned* pa = (const unsigned*)ldsA;
      int row = mt * 16 + ln;
#pragma unroll
      for (int p = 0; p < 8; ++p) {
        int kOff = 2 * p + hl * 16;  // {0,2,...,14}+16*hl
        af.d[p] = pa[row * 16 + (kOff >> 1)];
      }
      acc[mt][0] = __builtin_amdgcn_wmma_f32_16x16x32_bf16(false, af.bf, false, bf0.bf,
                                                           (short)0, acc[mt][0], false, false);
      acc[mt][1] = __builtin_amdgcn_wmma_f32_16x16x32_bf16(false, af.bf, false, bf1.bf,
                                                           (short)0, acc[mt][1], false, false);
    }
  }

#pragma unroll
  for (int mt = 0; mt < 8; ++mt) {
#pragma unroll
    for (int nt = 0; nt < 2; ++nt) {
      int c = (n0 + nt) * 16 + ln;
#pragma unroll
      for (int v = 0; v < 8; ++v) {
        int t = mbase + mt * 16 + v + hl * 8;
        if (t < TLEN) out[((size_t)b * TLEN + t) * DIMF + c] = acc[mt][nt][v];
      }
    }
  }
}

// ---------------------------------------------------------------------------
// Fully fused gating branch. block = (16-time tile) x batch; 8 waves.
// conv1 (g=2,k=3) -> LN -> gelu -> conv2 (g=2,k=3) -> LN -> sigmoid -> gate*.
// All intermediates live in LDS; halo recompute across tiles.
// ---------------------------------------------------------------------------
__global__ __launch_bounds__(256) void gated_conv(const float* __restrict__ x,
    const unsigned short* __restrict__ B1t, const unsigned short* __restrict__ B2t,
    const float* __restrict__ g1, const float* __restrict__ b1,
    const float* __restrict__ g2, const float* __restrict__ b2,
    float* __restrict__ out) {
  __shared__ float sGate[20][256];   // t0-2 .. t0+17, zero-padded
  __shared__ float sH1[18][128];     // h at t0-1 .. t0+16 (post LN+gelu)
  __shared__ float sH2[16][256];     // conv2 raw output
  const int b = blockIdx.y;
  const int t0 = blockIdx.x * 16;
  const int tid = threadIdx.x;
  const int w = tid >> 5, lane = tid & 31;
  const int hl = lane >> 4, ln = lane & 15;

  {   // phase 0: load gate halo
    const int c = tid;
#pragma unroll 4
    for (int r = 0; r < 20; ++r) {
      int t = t0 - 2 + r;
      sGate[r][c] = (t >= 0 && t < TLEN) ? x[((size_t)b * TLEN + t) * DIMF + CB + c] : 0.f;
    }
  }
  __syncthreads();

  {   // phase 1: conv1 as implicit GEMM. M rows m=0..17 valid (t=t0-1+m), N=128, K=384
    const int g = (w >= 4) ? 1 : 0;
    const int colBase = w * 16;        // output channel tile (group = o>=64)
    const int chBase = g * 128;        // gate channel base for this group
    v8f zero = {}; v8f acc0 = zero, acc1 = zero;
    for (int kc = 0; kc < 12; ++kc) {
      int kb = kc * 32;
      Frag bw;
      {
        const unsigned* pb = (const unsigned*)B1t;
        int o = colBase + ln;
        int base = (o * 384 + kb + hl * 16) >> 1;
#pragma unroll
        for (int p = 0; p < 8; ++p) bw.d[p] = pb[base + p];
      }
#pragma unroll
      for (int mt = 0; mt < 2; ++mt) {
        Frag af;
        int m = mt * 16 + ln;
#pragma unroll
        for (int j = 0; j < 16; ++j) {
          int kk = kb + hl * 16 + j;
          int i = kk / 3, k3 = kk - 3 * i;                 // kappa = i*3 + k
          float v = (m < 18) ? sGate[m + k3][chBase + i] : 0.f;
          af.us[j] = f2bf(v);
        }
        v8f& a = (mt == 0) ? acc0 : acc1;
        a = __builtin_amdgcn_wmma_f32_16x16x32_bf16(false, af.bf, false, bw.bf,
                                                    (short)0, a, false, false);
      }
    }
#pragma unroll
    for (int v = 0; v < 8; ++v) {
      int m0 = v + hl * 8;
      if (m0 < 18) sH1[m0][colBase + ln] = acc0[v];
      int m1 = 16 + v + hl * 8;
      if (m1 < 18) sH1[m1][colBase + ln] = acc1[v];
    }
  }
  __syncthreads();

  // phase 2: LayerNorm(128) + exact gelu, row per wave; invalid t rows -> 0 (conv2 zero pad)
  for (int r = w; r < 18; r += 8) {
    int t = t0 - 1 + r;
    float v[4];
#pragma unroll
    for (int q = 0; q < 4; ++q) v[q] = sH1[r][lane + 32 * q];
    float s = v[0] + v[1] + v[2] + v[3];
    s = wave_sum(s);
    float mean = s * (1.f / 128.f);
    float sq = 0.f;
#pragma unroll
    for (int q = 0; q < 4; ++q) { float d = v[q] - mean; sq += d * d; }
    sq = wave_sum(sq);
    float rstd = rsqrtf(sq * (1.f / 128.f) + 1e-5f);
    bool valid = (t >= 0 && t < TLEN);
#pragma unroll
    for (int q = 0; q < 4; ++q) {
      int c = lane + 32 * q;
      float y = (v[q] - mean) * rstd * g1[c] + b1[c];
      float ge = 0.5f * y * (1.f + erff(y * 0.70710678118654752f));
      sH1[r][c] = valid ? ge : 0.f;
    }
  }
  __syncthreads();

  {   // phase 3: conv2 as implicit GEMM. M=16 (t=t0+m), N=256, K=192
    const int nt0 = w * 2;
    const int hbase = (w >= 4) ? 64 : 0;   // group of o = [32w, 32w+32)
    v8f zero = {}; v8f acc0 = zero, acc1 = zero;
    for (int kc = 0; kc < 6; ++kc) {
      int kb = kc * 32;
      Frag af;
#pragma unroll
      for (int j = 0; j < 16; ++j) {
        int kk = kb + hl * 16 + j;
        int i = kk / 3, k3 = kk - 3 * i;
        af.us[j] = f2bf(sH1[ln + k3][hbase + i]);
      }
      Frag bw0, bw1;
      {
        const unsigned* pb = (const unsigned*)B2t;
        int o0 = nt0 * 16 + ln;
        int base0 = (o0 * 192 + kb + hl * 16) >> 1;
        int base1 = ((o0 + 16) * 192 + kb + hl * 16) >> 1;
#pragma unroll
        for (int p = 0; p < 8; ++p) { bw0.d[p] = pb[base0 + p]; bw1.d[p] = pb[base1 + p]; }
      }
      acc0 = __builtin_amdgcn_wmma_f32_16x16x32_bf16(false, af.bf, false, bw0.bf,
                                                     (short)0, acc0, false, false);
      acc1 = __builtin_amdgcn_wmma_f32_16x16x32_bf16(false, af.bf, false, bw1.bf,
                                                     (short)0, acc1, false, false);
    }
#pragma unroll
    for (int v = 0; v < 8; ++v) {
      int m = v + hl * 8;
      sH2[m][nt0 * 16 + ln] = acc0[v];
      sH2[m][(nt0 + 1) * 16 + ln] = acc1[v];
    }
  }
  __syncthreads();

  // phase 4: LayerNorm(256) + sigmoid + gate multiply + store (2 rows/wave)
#pragma unroll
  for (int rr = 0; rr < 2; ++rr) {
    int r = w * 2 + rr;
    int t = t0 + r;
    float v[8];
#pragma unroll
    for (int q = 0; q < 8; ++q) v[q] = sH2[r][lane + 32 * q];
    float s = 0.f;
#pragma unroll
    for (int q = 0; q < 8; ++q) s += v[q];
    s = wave_sum(s);
    float mean = s * (1.f / 256.f);
    float sq = 0.f;
#pragma unroll
    for (int q = 0; q < 8; ++q) { float d = v[q] - mean; sq += d * d; }
    sq = wave_sum(sq);
    float rstd = rsqrtf(sq * (1.f / 256.f) + 1e-5f);
    if (t < TLEN) {
#pragma unroll
      for (int q = 0; q < 8; ++q) {
        int c = lane + 32 * q;
        float y = (v[q] - mean) * rstd * g2[c] + b2[c];
        float sg = 1.f / (1.f + __expf(-y));
        out[((size_t)b * TLEN + t) * DIMF + CB + c] = sGate[r + 2][c] * sg;
      }
    }
  }
}

// ---------------------------------------------------------------------------
extern "C" void kernel_launch(void* const* d_in, const int* in_sizes, int n_in,
                              void* d_out, int out_size, void* d_ws, size_t ws_size,
                              hipStream_t stream) {
  (void)in_sizes; (void)n_in; (void)out_size; (void)ws_size;
  const float* x  = (const float*)d_in[0];
  const float* lf = (const float*)d_in[1];
  const float* w1 = (const float*)d_in[2];
  const float* w2 = (const float*)d_in[3];
  const float* g1 = (const float*)d_in[4];
  const float* b1 = (const float*)d_in[5];
  const float* g2 = (const float*)d_in[6];
  const float* b2 = (const float*)d_in[7];
  float* out = (float*)d_out;

  uint8_t* ws = (uint8_t*)d_ws;
  unsigned short* Mbf = (unsigned short*)(ws + 0);                       // 131072 B
  float* P            = (float*)(ws + 131072);                           // 237168 B
  unsigned short* B1t = (unsigned short*)(ws + 131072 + 237568);         //  98304 B
  unsigned short* B2t = (unsigned short*)(ws + 131072 + 237568 + 98304); //  98304 B

  build_P<<<(FREQ * TLEN + 255) / 256, 256, 0, stream>>>(lf, P);
  build_M<<<(256 * 256) / 256, 256, 0, stream>>>(P, Mbf);
  build_W<<<(128 * 384 + 256 * 192 + 255) / 256, 256, 0, stream>>>(w1, w2, B1t, B2t);
  fft_gemm<<<dim3(2, 512), 256, 0, stream>>>(x, Mbf, out);
  gated_conv<<<dim3(16, 512), 256, 0, stream>>>(x, B1t, B2t, g1, b1, g2, b2, out);
}
